// MultiHeadedAttention_2327872274798
// MI455X (gfx1250) — compile-verified, matrix operand order checked by reading, not run on hardware
//
#include <hip/hip_runtime.h>
#include <hip/hip_bf16.h>
#include <stdint.h>
#include <stddef.h>

// Problem constants (match reference)
#define BB 2
#define SS 2048
#define DD 1024
#define HH 16
#define DHD 64
#define MMR (BB * SS) // 4096 rows for the projection GEMMs

typedef __attribute__((ext_vector_type(16))) __bf16 v16bf;
typedef __attribute__((ext_vector_type(8)))  float  v8f;
typedef __attribute__((ext_vector_type(8)))  unsigned short ush8;
typedef __attribute__((ext_vector_type(16))) unsigned short ush16;

// f32 -> bf16 with round-to-nearest-even
__device__ __forceinline__ unsigned short f2bf(float f) {
  unsigned int u = __builtin_bit_cast(unsigned int, f);
  u += 0x7FFFu + ((u >> 16) & 1u);
  return (unsigned short)(u >> 16);
}

// Load a 16x32 bf16 WMMA fragment (A-pattern; also used for B when B is
// stored row-major as [N x K]) from an LDS tile.
// Per ISA 7.12.2: lanes 0-15 hold K=[h*8 .. h*8+7] in VGPRs 0-3 and
// K=[16+h*8 .. 16+h*8+7] in VGPRs 4-7 (h = lane>>4) => two 16B ds loads.
__device__ __forceinline__ v16bf load_frag_lds(const unsigned short* tile,
                                               int row, int k0, int stride) {
  const int lane = threadIdx.x & 31;
  const int m = lane & 15;
  const int h = lane >> 4;
  const unsigned short* p = tile + (size_t)(row + m) * stride + k0 + h * 8;
  ush8 lo = *(const ush8*)(p);
  ush8 hi = *(const ush8*)(p + 16);
  ush16 f;
#pragma unroll
  for (int i = 0; i < 8; ++i) { f[i] = lo[i]; f[i + 8] = hi[i]; }
  return __builtin_bit_cast(v16bf, f);
}

__device__ __forceinline__ v8f wmma_bf16(v16bf a, v16bf b, v8f c) {
  return __builtin_amdgcn_wmma_f32_16x16x32_bf16(false, a, false, b,
                                                 (short)0, c, false, false);
}

// ---------------------------------------------------------------------------
// Weight transpose + convert: Wt[n][k] = bf16(W[k][n]), 1024x1024
// ---------------------------------------------------------------------------
__global__ __launch_bounds__(256) void transpose_conv(
    const float* __restrict__ W, unsigned short* __restrict__ Wt) {
  __shared__ float tile[32][33];
  const int tx = threadIdx.x & 31;
  const int ty = threadIdx.x >> 5; // 0..7
  const int c0 = blockIdx.x * 32;
  const int r0 = blockIdx.y * 32;
#pragma unroll
  for (int i = 0; i < 4; ++i) {
    int r = ty + i * 8;
    tile[r][tx] = W[(size_t)(r0 + r) * DD + c0 + tx];
  }
  __syncthreads();
#pragma unroll
  for (int i = 0; i < 4; ++i) {
    int r = ty + i * 8;
    Wt[(size_t)(c0 + r) * DD + r0 + tx] = f2bf(tile[tx][r]);
  }
}

// ---------------------------------------------------------------------------
// Projection GEMM: out = A[4096x1024](f32) @ Bt[1024x1024]^T (bf16) + bias
// 256 threads = 8 waves; tile 128x128; each wave 32x64 (2x4 fragments).
// mode 0: K -> heads bf16 [B,H,S,DH]
// mode 1: Q -> heads bf16 [B,H,S,DH], scaled by 1/sqrt(DH)
// mode 2: V -> heads bf16 transposed [B,H,DH,S]
// mode 3: O -> f32 [B,S,D] (A is f32 context)
// ---------------------------------------------------------------------------
__global__ __launch_bounds__(256) void gemm_proj(
    const float* __restrict__ A, const unsigned short* __restrict__ Bt,
    const float* __restrict__ bias, unsigned short* __restrict__ outb,
    float* __restrict__ outf, int mode) {
  __shared__ __align__(16) unsigned short sA[128][40];
  __shared__ __align__(16) unsigned short sB[128][40];
  const int t = threadIdx.x;
  const int m0 = blockIdx.x * 128;
  const int n0 = blockIdx.y * 128;
  const int wid = t >> 5;
  const int wr = wid >> 1; // 0..3 (row group of 32)
  const int wc = wid & 1;  // 0..1 (col group of 64)

  v8f acc[2][4];
#pragma unroll
  for (int mi = 0; mi < 2; ++mi)
#pragma unroll
    for (int ni = 0; ni < 4; ++ni)
#pragma unroll
      for (int e = 0; e < 8; ++e) acc[mi][ni][e] = 0.0f;

  for (int k0 = 0; k0 < DD; k0 += 32) {
    // Stage A tile: 128 rows x 32 cols f32 -> bf16  (1024 float4 slots)
#pragma unroll
    for (int i = 0; i < 4; ++i) {
      int slot = t + 256 * i;
      int r = slot >> 3, q = slot & 7;
      float4 v = *(const float4*)(A + (size_t)(m0 + r) * DD + k0 + q * 4);
      sA[r][q * 4 + 0] = f2bf(v.x);
      sA[r][q * 4 + 1] = f2bf(v.y);
      sA[r][q * 4 + 2] = f2bf(v.z);
      sA[r][q * 4 + 3] = f2bf(v.w);
    }
    // Stage B tile: 128 rows(n) x 32 cols(k) bf16  (512 ush8 slots)
#pragma unroll
    for (int i = 0; i < 2; ++i) {
      int slot = t + 256 * i;
      int r = slot >> 2, q = slot & 3;
      *(ush8*)&sB[r][q * 8] =
          *(const ush8*)(Bt + (size_t)(n0 + r) * DD + k0 + q * 8);
    }
    __syncthreads();
    v16bf bfr[4];
#pragma unroll
    for (int ni = 0; ni < 4; ++ni)
      bfr[ni] = load_frag_lds(&sB[0][0], wc * 64 + ni * 16, 0, 40);
#pragma unroll
    for (int mi = 0; mi < 2; ++mi) {
      v16bf afr = load_frag_lds(&sA[0][0], wr * 32 + mi * 16, 0, 40);
#pragma unroll
      for (int ni = 0; ni < 4; ++ni)
        acc[mi][ni] = wmma_bf16(afr, bfr[ni], acc[mi][ni]);
    }
    __syncthreads();
  }

  // Epilogue: bias (+ optional scale) and scatter per mode.
  const int lane = t & 31;
  const int cl = lane & 15;
  const int rh = (lane >> 4) * 8;
#pragma unroll
  for (int mi = 0; mi < 2; ++mi)
#pragma unroll
    for (int ni = 0; ni < 4; ++ni) {
      int C = n0 + wc * 64 + ni * 16 + cl;
      float bv = bias[C];
#pragma unroll
      for (int e = 0; e < 8; ++e) {
        int R = m0 + wr * 32 + mi * 16 + rh + e;
        float v = acc[mi][ni][e] + bv;
        if (mode == 1) v *= 0.125f; // 1/sqrt(64)
        if (mode == 3) {
          outf[(size_t)R * DD + C] = v;
        } else {
          int b = R >> 11, s = R & 2047;
          int h = C >> 6, d = C & 63;
          if (mode == 2)
            outb[(((size_t)b * HH + h) * DHD + d) * SS + s] = f2bf(v);
          else
            outb[(((size_t)b * HH + h) * SS + s) * DHD + d] = f2bf(v);
        }
      }
    }
}

// ---------------------------------------------------------------------------
// Scores: for (b,h): S[q,k] = Q[q,:] . K[k,:]  (DH=64 contraction, 2 k-steps)
// 128 threads = 4 waves; 64x64 tile per workgroup. Raw f32 scores -> attn buf.
// ---------------------------------------------------------------------------
__global__ __launch_bounds__(128) void attn_scores(
    const unsigned short* __restrict__ qh, const unsigned short* __restrict__ kh,
    float* __restrict__ sc) {
  __shared__ __align__(16) unsigned short sQ[64][72];
  __shared__ __align__(16) unsigned short sK[64][72];
  const int t = threadIdx.x;
  const int k0 = blockIdx.x * 64;
  const int q0 = blockIdx.y * 64;
  const int bh = blockIdx.z;
  const unsigned short* qb = qh + ((size_t)bh * SS + q0) * DHD;
  const unsigned short* kb = kh + ((size_t)bh * SS + k0) * DHD;
#pragma unroll
  for (int i = 0; i < 4; ++i) {
    int slot = t + 128 * i; // 512 ush8 slots each
    int r = slot >> 3, q = slot & 7;
    *(ush8*)&sQ[r][q * 8] = *(const ush8*)(qb + (size_t)r * DHD + q * 8);
    *(ush8*)&sK[r][q * 8] = *(const ush8*)(kb + (size_t)r * DHD + q * 8);
  }
  __syncthreads();
  const int w = t >> 5;
  v8f acc[4];
#pragma unroll
  for (int nt = 0; nt < 4; ++nt)
#pragma unroll
    for (int e = 0; e < 8; ++e) acc[nt][e] = 0.0f;
#pragma unroll
  for (int ks = 0; ks < 2; ++ks) {
    v16bf a = load_frag_lds(&sQ[0][0], w * 16, ks * 32, 72);
#pragma unroll
    for (int nt = 0; nt < 4; ++nt) {
      v16bf b = load_frag_lds(&sK[0][0], nt * 16, ks * 32, 72);
      acc[nt] = wmma_bf16(a, b, acc[nt]);
    }
  }
  const int lane = t & 31, cl = lane & 15, rh = (lane >> 4) * 8;
  float* out = sc + (size_t)bh * SS * SS;
#pragma unroll
  for (int nt = 0; nt < 4; ++nt)
#pragma unroll
    for (int e = 0; e < 8; ++e)
      out[(size_t)(q0 + w * 16 + rh + e) * SS + k0 + nt * 16 + cl] = acc[nt][e];
}

// ---------------------------------------------------------------------------
// Row softmax with mask, in place. One block per row (B*H*S rows of length S).
// ---------------------------------------------------------------------------
__global__ __launch_bounds__(256) void softmax_rows(
    float* __restrict__ at, const unsigned char* __restrict__ mask) {
  __shared__ float red[256];
  const int t = threadIdx.x;
  const size_t row = blockIdx.x; // bh*S + sq
  const int bh = (int)(row >> 11);
  const int sq = (int)(row & 2047);
  const int b = bh >> 4;
  float* p = at + row * (size_t)SS;
  const unsigned char* mp = mask + ((size_t)b * SS + sq) * SS;
  float v[8];
  float mx = -3.4e38f;
#pragma unroll
  for (int j = 0; j < 8; ++j) {
    int c = t + 256 * j;
    float x = p[c];
    if (mp[c]) x = -1e10f;
    v[j] = x;
    mx = fmaxf(mx, x);
  }
  red[t] = mx;
  __syncthreads();
  for (int s = 128; s > 0; s >>= 1) {
    if (t < s) red[t] = fmaxf(red[t], red[t + s]);
    __syncthreads();
  }
  mx = red[0];
  __syncthreads();
  float sum = 0.0f;
#pragma unroll
  for (int j = 0; j < 8; ++j) { v[j] = __expf(v[j] - mx); sum += v[j]; }
  red[t] = sum;
  __syncthreads();
  for (int s = 128; s > 0; s >>= 1) {
    if (t < s) red[t] += red[t + s];
    __syncthreads();
  }
  const float inv = 1.0f / red[0];
#pragma unroll
  for (int j = 0; j < 8; ++j) p[t + 256 * j] = v[j] * inv;
}

// ---------------------------------------------------------------------------
// Context: for (b,h): C[q,d] = sum_k P[q,k] * V[k,d], with V stored transposed
// as vT[b,h,d,k] (bf16) so the B operand is a natural [N x K] row-major load.
// 128 threads = 4 waves; 64(q) x 64(d) tile; k loop over S in steps of 64.
// Output: f32 context [B,S,D] (head-merged), consumed by final projection.
// ---------------------------------------------------------------------------
__global__ __launch_bounds__(128) void attn_context(
    const float* __restrict__ at, const unsigned short* __restrict__ vT,
    float* __restrict__ ctx) {
  __shared__ __align__(16) unsigned short sP[64][72];
  __shared__ __align__(16) unsigned short sV[64][72];
  const int t = threadIdx.x;
  const int q0 = blockIdx.x * 64;
  const int bh = blockIdx.y;
  const int w = t >> 5;
  const float* pb = at + ((size_t)bh * SS + q0) * SS;
  const unsigned short* vb = vT + (size_t)bh * DHD * SS;
  v8f acc[4];
#pragma unroll
  for (int nt = 0; nt < 4; ++nt)
#pragma unroll
    for (int e = 0; e < 8; ++e) acc[nt][e] = 0.0f;

  for (int kk = 0; kk < SS; kk += 64) {
    // Stage P: 64 x 64 f32 -> bf16 (1024 float4 slots)
#pragma unroll
    for (int i = 0; i < 8; ++i) {
      int slot = t + 128 * i;
      int r = slot >> 4, q4 = slot & 15;
      float4 x = *(const float4*)(pb + (size_t)r * SS + kk + q4 * 4);
      sP[r][q4 * 4 + 0] = f2bf(x.x);
      sP[r][q4 * 4 + 1] = f2bf(x.y);
      sP[r][q4 * 4 + 2] = f2bf(x.z);
      sP[r][q4 * 4 + 3] = f2bf(x.w);
    }
    // Stage V^T: 64(d) x 64(k) bf16 (512 ush8 slots)
#pragma unroll
    for (int i = 0; i < 4; ++i) {
      int slot = t + 128 * i;
      int r = slot >> 3, q = slot & 7;
      *(ush8*)&sV[r][q * 8] = *(const ush8*)(vb + (size_t)r * SS + kk + q * 8);
    }
    __syncthreads();
#pragma unroll
    for (int ks = 0; ks < 2; ++ks) {
      v16bf a = load_frag_lds(&sP[0][0], w * 16, ks * 32, 72);
#pragma unroll
      for (int nt = 0; nt < 4; ++nt) {
        v16bf b = load_frag_lds(&sV[0][0], nt * 16, ks * 32, 72);
        acc[nt] = wmma_bf16(a, b, acc[nt]);
      }
    }
    __syncthreads();
  }
  const int lane = t & 31, cl = lane & 15, rh = (lane >> 4) * 8;
  const int b = bh >> 4, h = bh & 15;
#pragma unroll
  for (int nt = 0; nt < 4; ++nt)
#pragma unroll
    for (int e = 0; e < 8; ++e) {
      int srow = q0 + w * 16 + rh + e;
      ctx[((size_t)b * SS + srow) * DD + h * DHD + nt * 16 + cl] = acc[nt][e];
    }
}

// ---------------------------------------------------------------------------
// Host-side orchestration
// ---------------------------------------------------------------------------
extern "C" void kernel_launch(void* const* d_in, const int* in_sizes, int n_in,
                              void* d_out, int out_size, void* d_ws,
                              size_t ws_size, hipStream_t stream) {
  (void)in_sizes; (void)n_in; (void)out_size; (void)ws_size;
  // Inputs (setup_inputs order)
  const float* key   = (const float*)d_in[0];
  const float* value = (const float*)d_in[1];
  const float* query = (const float*)d_in[2];
  const unsigned char* mask = (const unsigned char*)d_in[3];
  const float* Wk = (const float*)d_in[4];
  const float* bk = (const float*)d_in[5];
  const float* Wv = (const float*)d_in[6];
  const float* bv = (const float*)d_in[7];
  const float* Wq = (const float*)d_in[8];
  const float* bq = (const float*)d_in[9];
  const float* Wo = (const float*)d_in[10];
  const float* bo = (const float*)d_in[11];

  // d_out: [output f32 (B*S*D)] ++ [attn f32 (B*H*S*S)]
  float* out_f = (float*)d_out;
  float* attn  = out_f + (size_t)BB * SS * DD;

  // Workspace layout (bytes): 4x 2MB bf16 weights, 3x 8MB bf16 heads, 16MB f32 ctx
  char* ws = (char*)d_ws;
  const size_t WSZ = (size_t)DD * DD * sizeof(unsigned short);      // 2 MiB
  const size_t HSZ = (size_t)BB * SS * DD * sizeof(unsigned short); // 8 MiB
  unsigned short* WkT = (unsigned short*)(ws);
  unsigned short* WvT = (unsigned short*)(ws + WSZ);
  unsigned short* WqT = (unsigned short*)(ws + 2 * WSZ);
  unsigned short* WoT = (unsigned short*)(ws + 3 * WSZ);
  unsigned short* khd = (unsigned short*)(ws + 4 * WSZ);
  unsigned short* qhd = (unsigned short*)(ws + 4 * WSZ + HSZ);
  unsigned short* vTd = (unsigned short*)(ws + 4 * WSZ + 2 * HSZ);
  float*          ctx = (float*)(ws + 4 * WSZ + 3 * HSZ);

  const dim3 tgrid(32, 32), tblk(256);
  hipLaunchKernelGGL(transpose_conv, tgrid, tblk, 0, stream, Wk, WkT);
  hipLaunchKernelGGL(transpose_conv, tgrid, tblk, 0, stream, Wv, WvT);
  hipLaunchKernelGGL(transpose_conv, tgrid, tblk, 0, stream, Wq, WqT);
  hipLaunchKernelGGL(transpose_conv, tgrid, tblk, 0, stream, Wo, WoT);

  const dim3 ggrid(MMR / 128, DD / 128), gblk(256);
  hipLaunchKernelGGL(gemm_proj, ggrid, gblk, 0, stream, key,   WkT, bk, khd, (float*)nullptr, 0);
  hipLaunchKernelGGL(gemm_proj, ggrid, gblk, 0, stream, value, WvT, bv, vTd, (float*)nullptr, 2);
  hipLaunchKernelGGL(gemm_proj, ggrid, gblk, 0, stream, query, WqT, bq, qhd, (float*)nullptr, 1);

  hipLaunchKernelGGL(attn_scores, dim3(SS / 64, SS / 64, BB * HH), dim3(128),
                     0, stream, qhd, khd, attn);

  hipLaunchKernelGGL(softmax_rows, dim3(BB * HH * SS), dim3(256), 0, stream,
                     attn, mask);

  hipLaunchKernelGGL(attn_context, dim3(SS / 64, BB * HH), dim3(128), 0,
                     stream, attn, vTd, ctx);

  hipLaunchKernelGGL(gemm_proj, ggrid, gblk, 0, stream, ctx, WoT, bo,
                     (unsigned short*)nullptr, out_f, 3);
}
